// VQCodebook_38285338476799
// MI455X (gfx1250) — matrix operand order repaired and checked by reading, not compiled, not run on hardware
//
#include <hip/hip_runtime.h>

typedef __attribute__((ext_vector_type(16))) _Float16 v16h;
typedef __attribute__((ext_vector_type(8)))  _Float16 v8h;
typedef __attribute__((ext_vector_type(8)))  float    v8f;
typedef __attribute__((ext_vector_type(4)))  float    v4f;

#define N_ROWS      65536   // B*M
#define DIM         64      // D
#define NCODES      1024    // K
#define ROWS_PER_WG 256     // 8 waves * 32 rows (2 x 16-row tiles per wave)
#define LDS_STRIDE  72      // halves per codebook row in LDS (144B, 36 dwords -> conflict-free)

// Load B fragments for code tile `cidx` from LDS and issue the 4 WMMAs for
// both 16-row tiles into ACC.  Also latches hn/code metadata for this stage.
#define COMPUTE_TILE(ACC, HN, CODE, cidx)                                      \
  {                                                                            \
    const int code_ = (cidx) * 16 + m;                                         \
    const _Float16* bp_ = sB + code_ * LDS_STRIDE + 8 * g;                     \
    v8h q0_ = *(const v8h*)(bp_);                                              \
    v8h q1_ = *(const v8h*)(bp_ + 16);                                         \
    v8h q2_ = *(const v8h*)(bp_ + 32);                                         \
    v8h q3_ = *(const v8h*)(bp_ + 48);                                         \
    v16h b0_, b1_;                                                             \
    _Pragma("unroll")                                                          \
    for (int j = 0; j < 8; ++j) {                                              \
      b0_[j] = q0_[j]; b0_[8 + j] = q1_[j];                                    \
      b1_[j] = q2_[j]; b1_[8 + j] = q3_[j];                                    \
    }                                                                          \
    HN = sHN[code_];                                                           \
    CODE = code_;                                                              \
    _Pragma("unroll")                                                          \
    for (int t = 0; t < 2; ++t) {                                              \
      v8f a_ = {};                                                             \
      a_ = __builtin_amdgcn_wmma_f32_16x16x32_f16(false, afrag[t][0], false,   \
                                                  b0_, (short)0, a_, false,    \
                                                  false);                      \
      a_ = __builtin_amdgcn_wmma_f32_16x16x32_f16(false, afrag[t][1], false,   \
                                                  b1_, (short)0, a_, false,    \
                                                  false);                      \
      ACC[t] = a_;                                                             \
    }                                                                          \
  }

// Argmax update for one pipelined stage.
#define SCORE_TILE(ACC, HN, CODE)                                              \
  {                                                                            \
    _Pragma("unroll")                                                          \
    for (int t = 0; t < 2; ++t) {                                              \
      _Pragma("unroll")                                                        \
      for (int r = 0; r < 8; ++r) {                                            \
        float v_ = ACC[t][r] - HN;                                             \
        if (v_ > bestV[t][r]) { bestV[t][r] = v_; bestI[t][r] = CODE; }        \
      }                                                                        \
    }                                                                          \
  }

// ---------------------------------------------------------------------------
// Kernel 1: nearest-code search.  score = x.e - 0.5||e||^2, argmax over codes.
// Dot products via v_wmma_f32_16x16x32_f16 (f16 in, f32 accumulate).
// Two 16-row tiles per wave share B fragments; 2-stage ping-pong accumulators
// so each WMMA group is followed by independent scoring VALU (covers the f16
// WMMA->VALU hazard).  __launch_bounds__(256,1): occupancy is LDS-bound
// (148KB/WG -> 2 WG/WGP), so let the allocator use VGPRs instead of spilling.
// ---------------------------------------------------------------------------
__global__ void __launch_bounds__(256, 1)
vq_argmin_kernel(const float* __restrict__ zE,
                 const float* __restrict__ embed,
                 int*   __restrict__ idxInt,
                 float* __restrict__ idxOutF) {
  extern __shared__ _Float16 smem[];
  _Float16* sB  = smem;                                   // [NCODES * LDS_STRIDE] f16 codebook
  float*    sHN = (float*)(smem + NCODES * LDS_STRIDE);   // [NCODES] 0.5*||e||^2

  const int tid = threadIdx.x;

  // ---- Stage codebook into LDS (f32 -> f16) + half-norms --------------------
  for (int r = tid; r < NCODES; r += blockDim.x) {
    const float* erow = embed + r * DIM;
    _Float16*    dst  = sB + r * LDS_STRIDE;
    float ss = 0.0f;
    #pragma unroll
    for (int d0 = 0; d0 < DIM; d0 += 8) {
      v4f f0 = *(const v4f*)(erow + d0);
      v4f f1 = *(const v4f*)(erow + d0 + 4);
      v8h h;
      #pragma unroll
      for (int j = 0; j < 4; ++j) {
        ss += f0[j] * f0[j] + f1[j] * f1[j];
        h[j]     = (_Float16)f0[j];
        h[4 + j] = (_Float16)f1[j];
      }
      *(v8h*)(dst + d0) = h;          // ds_store_b128, 16B aligned
    }
    sHN[r] = 0.5f * ss;
  }
  __syncthreads();

  // ---- Per-wave: two 16-row tiles ------------------------------------------
  const int lane   = tid & 31;
  const int waveId = tid >> 5;
  const int m      = lane & 15;       // A-matrix row within tile / B column
  const int g      = lane >> 4;       // lane half-group
  const int rowB0  = blockIdx.x * ROWS_PER_WG + waveId * 32;  // tile 0; tile 1 = +16

  // A fragments [tile][h]; ISA 16-bit A 16x32 layout:
  // lane group g, halves j: d = 32h + 8g + (j%8) + 16*(j/8)
  v16h afrag[2][2];
  #pragma unroll
  for (int t = 0; t < 2; ++t) {
    const float* zrow = zE + (size_t)(rowB0 + 16 * t + m) * DIM;
    #pragma unroll
    for (int h = 0; h < 2; ++h) {
      const float* p = zrow + 32 * h + 8 * g;
      v4f c0 = *(const v4f*)(p);
      v4f c1 = *(const v4f*)(p + 4);
      v4f c2 = *(const v4f*)(p + 16);
      v4f c3 = *(const v4f*)(p + 20);
      #pragma unroll
      for (int j = 0; j < 4; ++j) {
        afrag[t][h][j]      = (_Float16)c0[j];
        afrag[t][h][4 + j]  = (_Float16)c1[j];
        afrag[t][h][8 + j]  = (_Float16)c2[j];
        afrag[t][h][12 + j] = (_Float16)c3[j];
      }
    }
  }

  float bestV[2][8];
  int   bestI[2][8];
  #pragma unroll
  for (int t = 0; t < 2; ++t)
    #pragma unroll
    for (int r = 0; r < 8; ++r) { bestV[t][r] = -3.4e38f; bestI[t][r] = 0; }

  // ---- Scan all 64 code tiles, 2-stage software pipeline --------------------
  v8f accA[2], accB[2];
  float hnA, hnB;
  int   codeA, codeB;

  COMPUTE_TILE(accA, hnA, codeA, 0);
  COMPUTE_TILE(accB, hnB, codeB, 1);
  for (int c = 2; c < NCODES / 16; c += 2) {
    SCORE_TILE(accA, hnA, codeA);            // tile c-2 (independent of accB wmmas)
    COMPUTE_TILE(accA, hnA, codeA, c);       // tile c
    SCORE_TILE(accB, hnB, codeB);            // tile c-1 (covers accA wmma hazard)
    COMPUTE_TILE(accB, hnB, codeB, c + 1);   // tile c+1
  }
  SCORE_TILE(accA, hnA, codeA);              // tile 62
  SCORE_TILE(accB, hnB, codeB);              // tile 63

  // ---- Reduce argmax across the 16 lanes holding each row -------------------
  // C layout: lane l holds column n=l%16, rows m = 8*(l/16)+r  (r = VGPR idx)
  #pragma unroll
  for (int t = 0; t < 2; ++t) {
    #pragma unroll
    for (int r = 0; r < 8; ++r) {
      float v = bestV[t][r];
      int   i = bestI[t][r];
      #pragma unroll
      for (int off = 8; off >= 1; off >>= 1) {
        float ov = __shfl_xor(v, off, 32);
        int   oi = __shfl_xor(i, off, 32);
        if (ov > v || (ov == v && oi < i)) { v = ov; i = oi; }
      }
      if (m == 0) {
        const int row = rowB0 + 16 * t + 8 * g + r;
        idxInt[row]  = i;
        idxOutF[row] = (float)i;
      }
    }
  }
}

// ---------------------------------------------------------------------------
// Kernel 2: gather z_q = embed[idx] (exact f32), write straight-through output,
// and accumulate sum((z_q - z_e)^2) for the loss.
// ---------------------------------------------------------------------------
__global__ void vq_gather_loss_kernel(const float* __restrict__ zE,
                                      const float* __restrict__ embed,
                                      const int*   __restrict__ idxInt,
                                      float* __restrict__ zq,
                                      float* __restrict__ acc) {
  __shared__ float red[8];
  const int e    = blockIdx.x * blockDim.x + threadIdx.x;  // element id
  const int row  = e >> 6;
  const int d    = e & 63;
  const int code = idxInt[row];
  const float q  = embed[code * DIM + d];
  zq[e] = q;                               // z_q_st value == z_q
  float diff = q - zE[e];
  float p = diff * diff;
  #pragma unroll
  for (int off = 16; off >= 1; off >>= 1) p += __shfl_xor(p, off, 32);
  const int lane = threadIdx.x & 31;
  const int w    = threadIdx.x >> 5;
  if (lane == 0) red[w] = p;
  __syncthreads();
  if (threadIdx.x == 0) {
    float s = 0.0f;
    #pragma unroll
    for (int i = 0; i < 8; ++i) s += red[i];
    atomicAdd(acc, s);
  }
}

__global__ void vq_zero_kernel(float* acc) { *acc = 0.0f; }

__global__ void vq_finalize_kernel(const float* acc, float* loss) {
  // codebook_loss + 0.25*commitment_loss, both equal mean((z_q - z_e)^2)
  *loss = 1.25f * (*acc) / (float)((size_t)N_ROWS * DIM);
}

// ---------------------------------------------------------------------------
extern "C" void kernel_launch(void* const* d_in, const int* in_sizes, int n_in,
                              void* d_out, int out_size, void* d_ws, size_t ws_size,
                              hipStream_t stream) {
  const float* zE    = (const float*)d_in[0];   // (16,4096,64) f32
  const float* embed = (const float*)d_in[1];   // (1024,64) f32

  float* out  = (float*)d_out;
  float* zq   = out;                                  // N_ROWS*DIM
  float* idxF = out + (size_t)N_ROWS * DIM;           // N_ROWS (indices as float)
  float* loss = idxF + N_ROWS;                        // 1

  int*   idxInt = (int*)d_ws;                                   // N_ROWS ints
  float* acc    = (float*)((char*)d_ws + (size_t)N_ROWS * 4);   // 1 float

  const size_t ldsBytes = (size_t)NCODES * LDS_STRIDE * sizeof(_Float16)
                        + (size_t)NCODES * sizeof(float);

  vq_zero_kernel<<<1, 1, 0, stream>>>(acc);
  vq_argmin_kernel<<<N_ROWS / ROWS_PER_WG, 256, ldsBytes, stream>>>(zE, embed, idxInt, idxF);
  vq_gather_loss_kernel<<<(N_ROWS * DIM) / 256, 256, 0, stream>>>(zE, embed, idxInt, zq, acc);
  vq_finalize_kernel<<<1, 1, 0, stream>>>(acc, loss);
}
